// ViTPredictor_48404281425983
// MI455X (gfx1250) — matrix operand order repaired and checked by reading, not compiled, not run on hardware
//
#include <hip/hip_runtime.h>
#include <hip/hip_bf16.h>

// ---------------------------------------------------------------------------
// Types for CDNA5 WMMA (wave32): v16bf A/B fragments (8 VGPRs), v8f C/D.
// ---------------------------------------------------------------------------
typedef __bf16 v2bf  __attribute__((ext_vector_type(2)));
typedef __bf16 v4bf  __attribute__((ext_vector_type(4)));
typedef __bf16 v8bf  __attribute__((ext_vector_type(8)));
typedef __bf16 v16bf __attribute__((ext_vector_type(16)));
typedef float  v8f   __attribute__((ext_vector_type(8)));

// Types matching the async-LDS builtin's parameters (per hipcc diagnostic:
// pointer to "__attribute__((vector_size(16))) int" in AS1 / AS3).
typedef int v4i_vs __attribute__((vector_size(16)));
typedef __attribute__((address_space(1))) v4i_vs* gptr128_t;
typedef __attribute__((address_space(3))) v4i_vs* lptr128_t;

#define DEV_INLINE __device__ __forceinline__

// gfx1250 async global->LDS path (ASYNCcnt), guarded so compile never breaks.
#if __has_builtin(__builtin_amdgcn_global_load_async_to_lds_b128)
#define HAVE_ASYNC_LDS 1
#else
#define HAVE_ASYNC_LDS 0
#endif

DEV_INLINE void wait_async_zero() {
#if HAVE_ASYNC_LDS
#if __has_builtin(__builtin_amdgcn_s_wait_asynccnt)
  __builtin_amdgcn_s_wait_asynccnt(0);
#else
  asm volatile("s_wait_asynccnt 0" ::: "memory");
#endif
#endif
}

DEV_INLINE v16bf cat8(v8bf lo, v8bf hi) {
  return __builtin_shufflevector(lo, hi, 0,1,2,3,4,5,6,7,8,9,10,11,12,13,14,15);
}

// 16-bit A fragment (16x32, row-major in LDS, row stride 16B-aligned):
// lanes 0-15 hold row M=lane, K {0..7, 16..23}; lanes 16-31 row M=lane-16, K {8..15, 24..31}.
DEV_INLINE v16bf load_a_frag(const __bf16* row, int lane) {
  const int s = (lane & 16) ? 8 : 0;
  v8bf lo = *(const v8bf*)(row + s);
  v8bf hi = *(const v8bf*)(row + s + 16);
  return cat8(lo, hi);
}

// 16-bit B fragment (32x16) from K-contiguous (transposed) LDS storage:
// lanes 0-15 hold col N=lane, K {0..7, 16..23}; lanes 16-31 col N=lane-16, K {8..15, 24..31}.
DEV_INLINE v16bf load_b_frag(const __bf16* col, int lane) {
  const int t = (lane & 16) ? 8 : 0;
  v8bf lo = *(const v8bf*)(col + t);
  v8bf hi = *(const v8bf*)(col + t + 16);
  return cat8(lo, hi);
}

DEV_INLINE v8f wmma_bf16(v16bf a, v16bf b, v8f c) {
  // (neg_a, A, neg_b, B, c_mod, C, reuse_a, reuse_b)
  return __builtin_amdgcn_wmma_f32_16x16x32_bf16(false, a, false, b, (short)0, c, false, false);
}

// ---------------------------------------------------------------------------
// Weight prep: fp32 W[K,N] -> bf16 W^T[N,K]  (LDS-tiled, both sides coalesced).
// Run once per launch; weights are then streamed as pure bf16 copies in GEMMs.
// ---------------------------------------------------------------------------
__global__ __launch_bounds__(256) void transpose_bf16_kernel(
    const float* __restrict__ in, __bf16* __restrict__ out, int K, int N)
{
  __shared__ float tile[32][33];
  const int kt = blockIdx.x * 32, nt = blockIdx.y * 32;
  const int tid = threadIdx.x;
  const int r = tid >> 5, c = tid & 31;
  #pragma unroll
  for (int i = 0; i < 4; ++i)
    tile[r + i*8][c] = in[(size_t)(kt + r + i*8) * N + nt + c];
  __syncthreads();
  const int n  = tid >> 3;            // 0..31
  const int kc = (tid & 7) * 4;       // 0..28
  v4bf v;
  v[0] = (__bf16)tile[kc+0][n];
  v[1] = (__bf16)tile[kc+1][n];
  v[2] = (__bf16)tile[kc+2][n];
  v[3] = (__bf16)tile[kc+3][n];
  *(v4bf*)(out + (size_t)(nt + n) * K + kt + kc) = v;
}

// ---------------------------------------------------------------------------
// Tiled GEMM: C[M,N] = epilogue(A[M,K] @ B[K,N]), B given as bf16 B^T[N,K]
// (row stride ldb). A fp32 -> bf16 on stage; fp32 accumulate.
// 128x128x32 tiles, double-buffered LDS; 8 waves, each 32x64 (2x4 WMMA tiles).
// B^T tiles staged with async global->LDS (ASYNCcnt) when available.
// ACT: 0=none, 1=exact GELU. BIAS/RES/POS enable bias / residual / pos-emb.
// ---------------------------------------------------------------------------
template<int ACT, bool BIAS, bool RES, bool POS>
__global__ __launch_bounds__(256) void gemm_kernel(
    const float* __restrict__ A, const __bf16* __restrict__ BT,
    const float* __restrict__ bias, const float* res,
    const float* __restrict__ pos, float* C,
    int M, int N, int K, int ldb, int tmod)
{
  __shared__ __align__(16) __bf16 As[2][128][40];   // stride 80B (16B mult)
  __shared__ __align__(16) __bf16 Bs[2][128][40];   // K-contiguous: Bs[n][k]
  const int tid = threadIdx.x, lane = tid & 31, wid = tid >> 5;
  const int wm = wid & 3, wn = wid >> 2;
  const int m0 = blockIdx.y * 128, n0 = blockIdx.x * 128;
  const int ln15 = lane & 15;

  auto stage = [&](int buf, int k0) {
    #pragma unroll
    for (int i = 0; i < 4; ++i) {            // A tile 128x32, f32 -> bf16, b64 stores
      int idx = tid + i * 256;
      int r = idx >> 3, cc = (idx & 7) * 4;
      float4 f = *(const float4*)(A + (size_t)(m0 + r) * K + k0 + cc);
      v4bf v;
      v[0] = (__bf16)f.x; v[1] = (__bf16)f.y;
      v[2] = (__bf16)f.z; v[3] = (__bf16)f.w;
      *(v4bf*)&As[buf][r][cc] = v;
    }
    #pragma unroll
    for (int i = 0; i < 2; ++i) {            // B^T tile 128x32, raw bf16 16B copies
      int idx = tid + i * 256;
      int r = idx >> 2, cc = (idx & 3) * 8;
      const __bf16* gsrc = BT + (size_t)(n0 + r) * ldb + k0 + cc;
      __bf16* ldst = &Bs[buf][r][cc];
#if HAVE_ASYNC_LDS
      __builtin_amdgcn_global_load_async_to_lds_b128(
          (gptr128_t)gsrc, (lptr128_t)ldst, 0, 0);
#else
      *(v8bf*)ldst = *(const v8bf*)gsrc;
#endif
    }
  };

  v8f acc[2][4] = {};
  const int nsteps = K >> 5;
  stage(0, 0);
  for (int s = 0; s < nsteps; ++s) {
    wait_async_zero();                        // async B^T copies done before barrier
    __syncthreads();
    const int cur = s & 1;
    if (s + 1 < nsteps) stage(cur ^ 1, (s + 1) << 5);   // prefetch next tile
    v16bf a0 = load_a_frag(&As[cur][wm*32 +      ln15][0], lane);
    v16bf a1 = load_a_frag(&As[cur][wm*32 + 16 + ln15][0], lane);
    #pragma unroll
    for (int ni = 0; ni < 4; ++ni) {
      v16bf b = load_b_frag(&Bs[cur][wn*64 + ni*16 + ln15][0], lane);
      acc[0][ni] = wmma_bf16(a0, b, acc[0][ni]);
      acc[1][ni] = wmma_bf16(a1, b, acc[1][ni]);
    }
  }

  const int ho = ((lane >> 4) & 1) * 8;      // C/D layout: lanes 16-31 hold M=8+r
  #pragma unroll
  for (int mi = 0; mi < 2; ++mi) {
    #pragma unroll
    for (int ni = 0; ni < 4; ++ni) {
      #pragma unroll
      for (int r = 0; r < 8; ++r) {
        int row = m0 + wm*32 + mi*16 + ho + r;
        int col = n0 + wn*64 + ni*16 + ln15;
        float v = acc[mi][ni][r];
        if (BIAS) v += bias[col];
        if (RES)  v += res[(size_t)row * N + col];
        if (POS)  v += pos[(size_t)(row % tmod) * N + col];
        if (ACT == 1) v = 0.5f * v * (1.0f + erff(v * 0.70710678118654752f));
        C[(size_t)row * N + col] = v;
      }
    }
  }
}

// ---------------------------------------------------------------------------
// LayerNorm over H=512: one block (256 threads) per row.
// ---------------------------------------------------------------------------
__global__ __launch_bounds__(256) void ln_kernel(
    const float* __restrict__ x, const float* __restrict__ g,
    const float* __restrict__ b, float* __restrict__ out)
{
  __shared__ float r1[256], r2[256];
  const int row = blockIdx.x, tid = threadIdx.x;
  float a0 = x[(size_t)row*512 + tid];
  float a1 = x[(size_t)row*512 + tid + 256];
  r1[tid] = a0 + a1;
  r2[tid] = a0*a0 + a1*a1;
  __syncthreads();
  for (int s = 128; s > 0; s >>= 1) {
    if (tid < s) { r1[tid] += r1[tid+s]; r2[tid] += r2[tid+s]; }
    __syncthreads();
  }
  float mu  = r1[0] * (1.0f/512.0f);
  float var = r2[0] * (1.0f/512.0f) - mu*mu;
  float rs  = rsqrtf(var + 1e-5f);
  out[(size_t)row*512 + tid]       = (a0 - mu) * rs * g[tid]       + b[tid];
  out[(size_t)row*512 + tid + 256] = (a1 - mu) * rs * g[tid + 256] + b[tid + 256];
}

// ---------------------------------------------------------------------------
// Causal flash attention with WMMA. One block per (batch, head, 128-q tile).
// qkv: [B*T, 1536] with q|k|v concatenated; head h at column h*64.
// Each wave owns 16 query rows; K/V tiles of 128 keys staged cooperatively.
// ---------------------------------------------------------------------------
__global__ __launch_bounds__(256) void attn_kernel(const float* __restrict__ qkv,
                                                   float* o)
{
  __shared__ __align__(16) __bf16 Qs[128][72];       // Q tile (pre-scaled)
  __shared__ __align__(16) __bf16 Ks[128][72];       // K row-major == B^T layout
  __shared__ __align__(16) __bf16 Vt[64][136];       // V transposed: Vt[d][key]
  __shared__ __align__(16) __bf16 Ps[8][16][136];    // per-wave P round-trip

  const int tid = threadIdx.x, lane = tid & 31, wid = tid >> 5;
  const int qt = blockIdx.x & 7;           // T/128 = 8 query tiles
  const int bh = blockIdx.x >> 3;
  const int bb = bh >> 3, hh = bh & 7;     // NH = 8
  const int ho = ((lane >> 4) & 1) * 8;
  const int ln15 = lane & 15;
  const size_t rowbase = (size_t)bb * 1024;

  #pragma unroll
  for (int i = 0; i < 8; ++i) {            // stage Q, scaled by dh^-0.5 = 0.125
    int idx = tid + i*256;
    int r = idx >> 4, cc = (idx & 15) * 4;
    float4 f = *(const float4*)(qkv + (rowbase + qt*128 + r) * 1536 + hh*64 + cc);
    v4bf v;
    v[0] = (__bf16)(f.x * 0.125f); v[1] = (__bf16)(f.y * 0.125f);
    v[2] = (__bf16)(f.z * 0.125f); v[3] = (__bf16)(f.w * 0.125f);
    *(v4bf*)&Qs[r][cc] = v;
  }
  __syncthreads();

  v16bf aq0 = load_a_frag(&Qs[wid*16 + ln15][0],  lane);
  v16bf aq1 = load_a_frag(&Qs[wid*16 + ln15][32], lane);

  float mrow[8], lrow[8];
  v8f Oacc[4] = {};
  #pragma unroll
  for (int r = 0; r < 8; ++r) { mrow[r] = -1e30f; lrow[r] = 0.0f; }

  for (int kt = 0; kt <= qt; ++kt) {
    #pragma unroll
    for (int i = 0; i < 8; ++i) {          // stage K tile (packed b64 stores)
      int idx = tid + i*256;
      int r = idx >> 4, cc = (idx & 15) * 4;
      float4 f = *(const float4*)(qkv + (rowbase + kt*128 + r) * 1536 + 512 + hh*64 + cc);
      v4bf v;
      v[0] = (__bf16)f.x; v[1] = (__bf16)f.y;
      v[2] = (__bf16)f.z; v[3] = (__bf16)f.w;
      *(v4bf*)&Ks[r][cc] = v;
    }
    #pragma unroll
    for (int i = 0; i < 8; ++i) {          // stage V tile, transposed
      int idx = tid + i*256;
      int r = idx >> 4, cc = (idx & 15) * 4;
      float4 f = *(const float4*)(qkv + (rowbase + kt*128 + r) * 1536 + 1024 + hh*64 + cc);
      Vt[cc+0][r] = (__bf16)f.x; Vt[cc+1][r] = (__bf16)f.y;
      Vt[cc+2][r] = (__bf16)f.z; Vt[cc+3][r] = (__bf16)f.w;
    }
    __syncthreads();

    // S = (Q*scale) @ K^T : 16 queries x 128 keys per wave
    v8f s[8] = {};
    #pragma unroll
    for (int ni = 0; ni < 8; ++ni) {
      v16bf bk0 = load_b_frag(&Ks[ni*16 + ln15][0],  lane);
      s[ni] = wmma_bf16(aq0, bk0, s[ni]);
      v16bf bk1 = load_b_frag(&Ks[ni*16 + ln15][32], lane);
      s[ni] = wmma_bf16(aq1, bk1, s[ni]);
    }

    // causal mask + running row max
    float nm[8];
    #pragma unroll
    for (int r = 0; r < 8; ++r) nm[r] = mrow[r];
    #pragma unroll
    for (int ni = 0; ni < 8; ++ni) {
      int key = kt*128 + ni*16 + ln15;
      #pragma unroll
      for (int r = 0; r < 8; ++r) {
        int qg = qt*128 + wid*16 + ho + r;
        float sv = (key <= qg) ? s[ni][r] : -1e30f;
        s[ni][r] = sv;
        nm[r] = fmaxf(nm[r], sv);
      }
    }
    #pragma unroll
    for (int r = 0; r < 8; ++r) {          // reduce max over the 16-lane row group
      nm[r] = fmaxf(nm[r], __shfl_xor(nm[r], 1, 32));
      nm[r] = fmaxf(nm[r], __shfl_xor(nm[r], 2, 32));
      nm[r] = fmaxf(nm[r], __shfl_xor(nm[r], 4, 32));
      nm[r] = fmaxf(nm[r], __shfl_xor(nm[r], 8, 32));
      float fs = expf(mrow[r] - nm[r]);
      mrow[r] = nm[r];
      lrow[r] *= fs;
      #pragma unroll
      for (int j = 0; j < 4; ++j) Oacc[j][r] *= fs;
    }
    #pragma unroll
    for (int ni = 0; ni < 8; ++ni) {       // P = exp(S - m); C-layout -> LDS
      #pragma unroll
      for (int r = 0; r < 8; ++r) {
        float p = expf(s[ni][r] - mrow[r]);
        lrow[r] += p;
        Ps[wid][ho + r][ni*16 + ln15] = (__bf16)p;
      }
    }
    // O += P @ V (wave-private Ps; in-wave DS ordering guarantees RAW)
    #pragma unroll
    for (int ks = 0; ks < 4; ++ks) {
      v16bf ap = load_a_frag(&Ps[wid][ln15][ks*32], lane);
      #pragma unroll
      for (int nj = 0; nj < 4; ++nj) {
        v16bf bv = load_b_frag(&Vt[nj*16 + ln15][ks*32], lane);
        Oacc[nj] = wmma_bf16(ap, bv, Oacc[nj]);
      }
    }
    __syncthreads();
  }

  #pragma unroll
  for (int r = 0; r < 8; ++r) {
    float l = lrow[r];
    l += __shfl_xor(l, 1, 32);
    l += __shfl_xor(l, 2, 32);
    l += __shfl_xor(l, 4, 32);
    l += __shfl_xor(l, 8, 32);
    float inv = 1.0f / l;
    size_t row = rowbase + qt*128 + wid*16 + ho + r;
    #pragma unroll
    for (int nj = 0; nj < 4; ++nj)
      o[row * 512 + hh*64 + nj*16 + ln15] = Oacc[nj][r] * inv;
  }
}

// ---------------------------------------------------------------------------
// Orchestration
// ---------------------------------------------------------------------------
extern "C" void kernel_launch(void* const* d_in, const int* in_sizes, int n_in,
                              void* d_out, int out_size, void* d_ws, size_t ws_size,
                              hipStream_t stream) {
  const float* states   = (const float*)d_in[0];
  const float* actions  = (const float*)d_in[1];
  const float* W_state  = (const float*)d_in[2];
  const float* b_state  = (const float*)d_in[3];
  const float* W_action = (const float*)d_in[4];
  const float* b_action = (const float*)d_in[5];
  const float* W_comb   = (const float*)d_in[6];
  const float* b_comb   = (const float*)d_in[7];
  const float* pos_emb  = (const float*)d_in[8];
  const float* ln1_g    = (const float*)d_in[9];
  const float* ln1_b    = (const float*)d_in[10];
  const float* W_qkv    = (const float*)d_in[11];
  const float* W_o      = (const float*)d_in[12];
  const float* b_o      = (const float*)d_in[13];
  const float* ln2_g    = (const float*)d_in[14];
  const float* ln2_b    = (const float*)d_in[15];
  const float* W_ff1    = (const float*)d_in[16];
  const float* b_ff1    = (const float*)d_in[17];
  const float* W_ff2    = (const float*)d_in[18];
  const float* b_ff2    = (const float*)d_in[19];
  const float* lnf_g    = (const float*)d_in[20];
  const float* lnf_b    = (const float*)d_in[21];
  const float* W_out    = (const float*)d_in[22];
  const float* b_out    = (const float*)d_in[23];
  float* out = (float*)d_out;

  const int BT = 8192;                        // B*T
  float* x    = (float*)d_ws;                 // [8192,512]  residual stream
  float* hbuf = x    + (size_t)BT*512;        // [8192,512]  LN out / se
  float* obuf = hbuf + (size_t)BT*512;        // [8192,512]  ae / attn out
  float* qbuf = obuf + (size_t)BT*512;        // [8192,2048] qkv / ff

  // bf16 transposed-weight cache, after the fp32 buffers (16B aligned)
  __bf16* wt = (__bf16*)(qbuf + (size_t)BT*2048);
  __bf16* wt_state  = wt;                               // [512][512]
  __bf16* wt_action = wt_state  + (size_t)512*512;      // [512][64]
  __bf16* wt_comb   = wt_action + (size_t)512*64;       // [512][1024]
  __bf16* wt_qkv    = wt_comb   + (size_t)512*1024;     // 6 x [1536][512]
  __bf16* wt_o      = wt_qkv    + (size_t)6*1536*512;   // 6 x [512][512]
  __bf16* wt_ff1    = wt_o      + (size_t)6*512*512;    // 6 x [2048][512]
  __bf16* wt_ff2    = wt_ff1    + (size_t)6*2048*512;   // 6 x [512][2048]
  __bf16* wt_out    = wt_ff2    + (size_t)6*512*2048;   // [512][512]

  dim3 blk(256);
  auto xpose = [&](const float* w, __bf16* dst, int K, int N) {
    transpose_bf16_kernel<<<dim3(K/32, N/32), blk, 0, stream>>>(w, dst, K, N);
  };

  // ---- weight prep: fp32 [K,N] -> bf16 [N,K] ----
  xpose(W_state,  wt_state,  512, 512);
  xpose(W_action, wt_action, 64,  512);
  xpose(W_comb,   wt_comb,   1024, 512);
  for (int d = 0; d < 6; ++d) {
    xpose(W_qkv + (size_t)d*512*1536, wt_qkv + (size_t)d*1536*512, 512, 1536);
    xpose(W_o   + (size_t)d*512*512,  wt_o   + (size_t)d*512*512,  512, 512);
    xpose(W_ff1 + (size_t)d*512*2048, wt_ff1 + (size_t)d*2048*512, 512, 2048);
    xpose(W_ff2 + (size_t)d*2048*512, wt_ff2 + (size_t)d*512*2048, 2048, 512);
  }
  xpose(W_out, wt_out, 512, 512);

  dim3 g512(4, 64), g1536(12, 64), g2048(16, 64);

  // embeddings: se, ae, then x = se@Wc_top + ae@Wc_bot + b_comb + pos_emb
  gemm_kernel<0,true ,false,false><<<g512, blk, 0, stream>>>(states,  wt_state,  b_state,  nullptr, nullptr, hbuf, BT, 512, 512, 512, 1024);
  gemm_kernel<0,true ,false,false><<<g512, blk, 0, stream>>>(actions, wt_action, b_action, nullptr, nullptr, obuf, BT, 512, 64,  64,  1024);
  gemm_kernel<0,true ,false,false><<<g512, blk, 0, stream>>>(hbuf, wt_comb,       b_comb, nullptr, nullptr, x, BT, 512, 512, 1024, 1024);
  gemm_kernel<0,false,true ,true ><<<g512, blk, 0, stream>>>(obuf, wt_comb + 512, nullptr, x,      pos_emb, x, BT, 512, 512, 1024, 1024);

  for (int d = 0; d < 6; ++d) {
    ln_kernel<<<BT, blk, 0, stream>>>(x, ln1_g + d*512, ln1_b + d*512, hbuf);
    gemm_kernel<0,false,false,false><<<g1536, blk, 0, stream>>>(hbuf, wt_qkv + (size_t)d*1536*512, nullptr, nullptr, nullptr, qbuf, BT, 1536, 512, 512, 1024);
    attn_kernel<<<dim3(512), blk, 0, stream>>>(qbuf, obuf);
    gemm_kernel<0,true ,true ,false><<<g512, blk, 0, stream>>>(obuf, wt_o + (size_t)d*512*512, b_o + d*512, x, nullptr, x, BT, 512, 512, 512, 1024);
    ln_kernel<<<BT, blk, 0, stream>>>(x, ln2_g + d*512, ln2_b + d*512, hbuf);
    gemm_kernel<1,true ,false,false><<<g2048, blk, 0, stream>>>(hbuf, wt_ff1 + (size_t)d*2048*512, b_ff1 + d*2048, nullptr, nullptr, qbuf, BT, 2048, 512, 512, 1024);
    gemm_kernel<0,true ,true ,false><<<g512,  blk, 0, stream>>>(qbuf, wt_ff2 + (size_t)d*512*2048, b_ff2 + d*512,  x, nullptr, x, BT, 512, 2048, 2048, 1024);
  }

  ln_kernel<<<BT, blk, 0, stream>>>(x, lnf_g, lnf_b, hbuf);
  gemm_kernel<0,true,false,false><<<g512, blk, 0, stream>>>(hbuf, wt_out, b_out, nullptr, nullptr, out, BT, 512, 512, 512, 1024);

  (void)in_sizes; (void)n_in; (void)out_size; (void)ws_size;
}
